// SemanticRQVAE_48318382080354
// MI455X (gfx1250) — compile-verified, matrix operand order checked by reading, not run on hardware
//
#include <hip/hip_runtime.h>
#include <hip/hip_bf16.h>

typedef __attribute__((ext_vector_type(16))) __bf16 v16bf;
typedef __attribute__((ext_vector_type(8)))  float  v8f;

#define BDIM 65536
#define EDIM 384
#define HDIM 512
#define KCODES 256
#define QLVL 4

__device__ inline v8f wmma_bf16f32(v16bf a, v16bf b, v8f c) {
  return __builtin_amdgcn_wmma_f32_16x16x32_bf16(false, a, false, b, (short)0, c,
                                                 false, false);
}

// ---- CDNA5 async global->LDS copy (ASYNCcnt) -------------------------------
// Per-lane 16B: LDS[vdst] = MEM[vaddr]  (VGLOBAL GV mode, saddr = off)
__device__ inline void async_ld_b128(void* lds_ptr, const void* gptr) {
  unsigned lds_off = (unsigned)(size_t)lds_ptr;  // flat LDS addr: low 32 = offset
  unsigned long long ga = (unsigned long long)(size_t)gptr;
  asm volatile("global_load_async_to_lds_b128 %0, %1, off"
               :: "v"(lds_off), "v"(ga) : "memory");
}
__device__ inline void wait_async0() {  // all async copies done
  asm volatile("s_wait_asynccnt 0" ::: "memory");
}
__device__ inline void wait_async2() {  // allow 2 newest (next tile) in flight
  asm volatile("s_wait_asynccnt 2" ::: "memory");
}

// ---- fragment loaders -------------------------------------------------------
// 16-bit A/B fragment (ISA 7.12.2): lane L -> row/col = L&15, kBase=(L>>4)*8,
// elems 0..7 = k kBase.., elems 8..15 = k 16+kBase..
__device__ inline v16bf lds_frag(const __bf16* rowptr, int kBase) {
  const unsigned int* p = (const unsigned int*)rowptr;
  union { unsigned int u[8]; v16bf v; } r;
  int d = kBase >> 1;
#pragma unroll
  for (int e = 0; e < 4; ++e) {
    r.u[e]     = p[d + e];
    r.u[4 + e] = p[8 + d + e];
  }
  return r.v;
}
__device__ inline v16bf gmem_frag(const __bf16* p) {
  union { uint4 u[2]; v16bf v; } r;
  r.u[0] = *(const uint4*)(p);
  r.u[1] = *(const uint4*)(p + 16);
  return r.v;
}
__device__ inline v16bf f32_frag(const float* rp) {
  union { __bf16 b[16]; v16bf v; } r;
#pragma unroll
  for (int e = 0; e < 8; ++e) {
    r.b[e]     = (__bf16)rp[e];
    r.b[8 + e] = (__bf16)rp[16 + e];
  }
  return r.v;
}

// ---- small helper kernels ---------------------------------------------------
__global__ void cvt_bf16_kernel(const float* __restrict__ s, __bf16* __restrict__ d, int n) {
  int i = blockIdx.x * blockDim.x + threadIdx.x;
  if (i < n) d[i] = (__bf16)s[i];
}

// transpose-convert: d[n][k] = (bf16)s[k][n]
__global__ void cvtT_bf16_kernel(const float* __restrict__ s, __bf16* __restrict__ d,
                                 int K, int N) {
  int k = blockIdx.x * 32 + (threadIdx.x & 31);
  int n = blockIdx.y * 8 + (threadIdx.x >> 5);
  if (k < K && n < N) d[(size_t)n * K + k] = (__bf16)s[(size_t)k * N + n];
}

__global__ void cnorm_kernel(const float* __restrict__ cb, float* __restrict__ cn) {
  int b = blockIdx.x;
  int lane = threadIdx.x;  // 32
  float s = 0.f;
  for (int i = lane; i < HDIM; i += 32) {
    float v = cb[(size_t)b * HDIM + i];
    s += v * v;
  }
#pragma unroll
  for (int mm = 16; mm >= 1; mm >>= 1) s += __shfl_xor(s, mm, 32);
  if (lane == 0) cn[b] = s;
}

__global__ void finalize_kernel(const float* __restrict__ reconPart, int nr,
                                const float* __restrict__ commitPart, int nc,
                                float* __restrict__ tail) {
  if (blockIdx.x == 0 && threadIdx.x == 0) {
    double rs = 0.0, cs = 0.0;
    for (int i = 0; i < nr; ++i) rs += (double)reconPart[i];
    for (int i = 0; i < nc; ++i) cs += (double)commitPart[i];
    tail[0] = (float)(rs / ((double)BDIM * (double)EDIM));
    tail[1] = (float)(0.25 * cs / ((double)BDIM * (double)HDIM));
  }
}

// ---- tiled bf16 WMMA GEMM: LDS-resident B panel + double-buffered A --------
// C[M,N] = epi(A[M,KD] * B[KD,N] + bias); Bt is pre-transposed [N][KD] bf16.
// EPI 0: relu->bf16, 1: f32, 2: f32 + (out-x)^2 block partials
#define AST 40  // A-tile LDS stride (elems): 80B rows, 16B-aligned for B128

__device__ inline void loadA4(const float* s, __bf16* d) {
  float4 v = *(const float4*)s;
  d[0] = (__bf16)v.x; d[1] = (__bf16)v.y; d[2] = (__bf16)v.z; d[3] = (__bf16)v.w;
}

template <typename AT, int EPI, int KD>
__global__ __launch_bounds__(256)
void gemm_kernel(const AT* __restrict__ A, const __bf16* __restrict__ Bt,
                 const float* __restrict__ bias, void* __restrict__ outp,
                 const float* __restrict__ Xref, float* __restrict__ lossPart,
                 int M, int N) {
  constexpr int KST = KD + 8;   // B panel row stride (elems); 16B-aligned rows
  constexpr int CPR = KD >> 3;  // 16B chunks per B row
  constexpr int NT  = KD / 32;  // k-tiles
  extern __shared__ __bf16 sh[];
  __bf16* Bp  = sh;                    // [128][KST]
  __bf16* As0 = sh + 128 * KST;        // [128][AST] ping
  __bf16* As1 = As0 + 128 * AST;       // [128][AST] pong
  float*  red = (float*)(As1 + 128 * AST);

  const int tid = threadIdx.x;
  const int wave = tid >> 5, lane = tid & 31, lo = lane & 15, hi = lane >> 4;
  const int m0 = blockIdx.y * 128, n0 = blockIdx.x * 128;
  const int rg = wave >> 1;  // 32-row group
  const int cg = wave & 1;   // 64-col group

  // ---- stage whole B panel once (async copy, padded stride) ----
  for (int s = tid; s < 128 * CPR; s += 256) {
    int n = s / CPR, kc = (s % CPR) << 3;
    async_ld_b128(&Bp[n * KST + kc], Bt + (size_t)(n0 + n) * KD + kc);
  }

  v8f acc[2][4];
#pragma unroll
  for (int i = 0; i < 2; ++i)
#pragma unroll
    for (int j = 0; j < 4; ++j)
#pragma unroll
      for (int r = 0; r < 8; ++r) acc[i][j][r] = 0.f;

  if constexpr (sizeof(AT) == 2) {
    // ---- software-pipelined: async-copy tile kt+1 while computing tile kt ----
    // (async loads retire in order; 2 async instrs per wave per tile)
    const __bf16* Ab = (const __bf16*)A;
    {
      int m = tid >> 2, kc = (tid & 3) << 3;  // tile 0, chunk 0/1
      async_ld_b128(&As0[m * AST + kc], Ab + (size_t)(m0 + m) * KD + kc);
      int s = 256 + tid;
      int m2 = s >> 2, kc2 = (s & 3) << 3;
      async_ld_b128(&As0[m2 * AST + kc2], Ab + (size_t)(m0 + m2) * KD + kc2);
    }
    for (int kt = 0; kt < NT; ++kt) {
      __bf16* cur = (kt & 1) ? As1 : As0;
      __bf16* nxt = (kt & 1) ? As0 : As1;
      if (kt + 1 < NT) {
        int k0n = (kt + 1) * 32;
#pragma unroll
        for (int it = 0; it < 2; ++it) {
          int s = it * 256 + tid;
          int m = s >> 2, kc = (s & 3) << 3;
          async_ld_b128(&nxt[m * AST + kc],
                        Ab + (size_t)(m0 + m) * KD + k0n + kc);
        }
        wait_async2();  // tile kt (and B panel) complete; tile kt+1 in flight
      } else {
        wait_async0();
      }
      __syncthreads();

      v16bf af[2];
#pragma unroll
      for (int i = 0; i < 2; ++i)
        af[i] = lds_frag(&cur[(rg * 32 + i * 16 + lo) * AST], hi * 8);
#pragma unroll
      for (int j = 0; j < 4; ++j) {
        v16bf bf = lds_frag(&Bp[(cg * 64 + j * 16 + lo) * KST + kt * 32], hi * 8);
#pragma unroll
        for (int i = 0; i < 2; ++i) acc[i][j] = wmma_bf16f32(af[i], bf, acc[i][j]);
      }
      __syncthreads();
    }
  } else {
    // ---- f32 A: load + convert + DS store per tile ----
    for (int kt = 0; kt < NT; ++kt) {
      int k0 = kt * 32;
#pragma unroll
      for (int it = 0; it < 4; ++it) {
        int s = it * 256 + tid;
        int m = s >> 3, kq = (s & 7) << 2;
        loadA4((const float*)A + (size_t)(m0 + m) * KD + k0 + kq,
               &As0[m * AST + kq]);
      }
      if (kt == 0) wait_async0();  // B panel
      __syncthreads();

      v16bf af[2];
#pragma unroll
      for (int i = 0; i < 2; ++i)
        af[i] = lds_frag(&As0[(rg * 32 + i * 16 + lo) * AST], hi * 8);
#pragma unroll
      for (int j = 0; j < 4; ++j) {
        v16bf bf = lds_frag(&Bp[(cg * 64 + j * 16 + lo) * KST + k0], hi * 8);
#pragma unroll
        for (int i = 0; i < 2; ++i) acc[i][j] = wmma_bf16f32(af[i], bf, acc[i][j]);
      }
      __syncthreads();
    }
  }

  float lsum = 0.f;
#pragma unroll
  for (int i = 0; i < 2; ++i) {
#pragma unroll
    for (int j = 0; j < 4; ++j) {
      int gn = n0 + cg * 64 + j * 16 + lo;
      float bv = bias[gn];
      int mb = m0 + rg * 32 + i * 16 + hi * 8;
#pragma unroll
      for (int r = 0; r < 8; ++r) {
        float v = acc[i][j][r] + bv;
        size_t off = (size_t)(mb + r) * N + gn;
        if constexpr (EPI == 0) {
          ((__bf16*)outp)[off] = (__bf16)fmaxf(v, 0.f);
        } else if constexpr (EPI == 1) {
          ((float*)outp)[off] = v;
        } else {
          ((float*)outp)[off] = v;
          float d = v - Xref[off];
          lsum += d * d;
        }
      }
    }
  }
  if constexpr (EPI == 2) {
#pragma unroll
    for (int mm = 16; mm >= 1; mm >>= 1) lsum += __shfl_xor(lsum, mm, 32);
    if (lane == 0) red[wave] = lsum;
    __syncthreads();
    if (tid == 0) {
      float s = 0.f;
      for (int w = 0; w < 8; ++w) s += red[w];
      lossPart[blockIdx.y * gridDim.x + blockIdx.x] = s;
    }
  }
}

// ---- Residual VQ kernel -----------------------------------------------------
// 128 rows/block, 8 waves; fp32 residual LDS-resident (258 KB of the 320 KB/WGP).
// A-fragments hoisted per level (128 VGPRs) and reused across all 16 code tiles.
#define RSTRIDE 516

__global__ __launch_bounds__(256)
void rvq_kernel(const float* __restrict__ z, const __bf16* __restrict__ cbb,
                const float* __restrict__ cbf, const float* __restrict__ cnorm,
                __bf16* __restrict__ qz, float* __restrict__ idx_out,
                float* __restrict__ commitPart) {
  extern __shared__ float smem[];
  float* res   = smem;                 // 128 * 516
  float* sc    = res + 128 * RSTRIDE;  // 8 * 272
  float* bestd = sc + 8 * 272;         // 128
  int*   besti = (int*)(bestd + 128);  // 128

  const int tid = threadIdx.x;
  const int wave = tid >> 5, lane = tid & 31, lo = lane & 15, hi = lane >> 4;
  const int m0 = blockIdx.x * 128;
  const int rowbase = wave * 16;

  for (int it = 0; it < 64; ++it) {
    int s = it * 256 + tid;
    int m = s >> 7, c = (s & 127) << 2;
    float4 v = *(const float4*)(z + (size_t)(m0 + m) * HDIM + c);
    float* rp = res + m * RSTRIDE + c;
    rp[0] = v.x; rp[1] = v.y; rp[2] = v.z; rp[3] = v.w;
  }
  __syncthreads();

  float lsum = 0.f;
  const float* aptr = res + (size_t)(rowbase + lo) * RSTRIDE + hi * 8;
  for (int q = 0; q < QLVL; ++q) {
    if (lane < 16) { bestd[rowbase + lane] = 3.4e38f; besti[rowbase + lane] = 0; }
    const __bf16* cbq = cbb + (size_t)q * KCODES * HDIM;
    const float*  cnq = cnorm + q * KCODES;

    // hoist all 16 A-fragments of this wave's 16 rows for the whole level
    v16bf af[16];
#pragma unroll
    for (int ks = 0; ks < 16; ++ks) af[ks] = f32_frag(aptr + ks * 32);

    for (int t = 0; t < 16; ++t) {
      v8f acc;
#pragma unroll
      for (int r = 0; r < 8; ++r) acc[r] = 0.f;
      const __bf16* bptr = cbq + (size_t)(t * 16 + lo) * HDIM + hi * 8;
#pragma unroll
      for (int ks = 0; ks < 16; ++ks)
        acc = wmma_bf16f32(af[ks], gmem_frag(bptr + ks * 32), acc);

      float* scw = sc + wave * 272;
#pragma unroll
      for (int r = 0; r < 8; ++r) scw[(r + 8 * hi) * 17 + lo] = acc[r];
      if (lane < 16) {
        float bd = bestd[rowbase + lane];
        int bi = besti[rowbase + lane];
#pragma unroll
        for (int c = 0; c < 16; ++c) {
          float d = cnq[t * 16 + c] - 2.f * scw[lane * 17 + c];
          if (d < bd) { bd = d; bi = t * 16 + c; }
        }
        bestd[rowbase + lane] = bd;
        besti[rowbase + lane] = bi;
      }
    }
    if (lane < 16)
      idx_out[(size_t)(m0 + rowbase + lane) * QLVL + q] =
          (float)besti[rowbase + lane];

    // fp32 residual update (2 lanes/row); commit term = sum(new_residual^2)
    {
      int r = rowbase + (lane >> 1);
      int ci = besti[r];
      const float* cv = cbf + ((size_t)q * KCODES + ci) * HDIM + (lane & 1) * 256;
      float* rr = res + (size_t)r * RSTRIDE + (lane & 1) * 256;
      for (int c = 0; c < 256; c += 4) {
        float4 w = *(const float4*)(cv + c);
        float n0 = rr[c + 0] - w.x; rr[c + 0] = n0; lsum += n0 * n0;
        float n1 = rr[c + 1] - w.y; rr[c + 1] = n1; lsum += n1 * n1;
        float n2 = rr[c + 2] - w.z; rr[c + 2] = n2; lsum += n2 * n2;
        float n3 = rr[c + 3] - w.w; rr[c + 3] = n3; lsum += n3 * n3;
      }
    }
  }
  __syncthreads();

  // quantized = z - residual_final -> bf16
  for (int it = 0; it < 64; ++it) {
    int s = it * 256 + tid;
    int m = s >> 7, c = (s & 127) << 2;
    float4 v = *(const float4*)(z + (size_t)(m0 + m) * HDIM + c);
    const float* rp = res + m * RSTRIDE + c;
    __bf16* d = qz + (size_t)(m0 + m) * HDIM + c;
    d[0] = (__bf16)(v.x - rp[0]);
    d[1] = (__bf16)(v.y - rp[1]);
    d[2] = (__bf16)(v.z - rp[2]);
    d[3] = (__bf16)(v.w - rp[3]);
  }

#pragma unroll
  for (int mm = 16; mm >= 1; mm >>= 1) lsum += __shfl_xor(lsum, mm, 32);
  __syncthreads();
  if (lane == 0) sc[wave] = lsum;
  __syncthreads();
  if (tid == 0) {
    float s = 0.f;
    for (int w = 0; w < 8; ++w) s += sc[w];
    commitPart[blockIdx.x] = s;
  }
}

// ---- host launcher ----------------------------------------------------------
extern "C" void kernel_launch(void* const* d_in, const int* in_sizes, int n_in,
                              void* d_out, int out_size, void* d_ws, size_t ws_size,
                              hipStream_t stream) {
  (void)in_sizes; (void)n_in; (void)out_size; (void)ws_size;
  const float* x   = (const float*)d_in[0];
  const float* W1  = (const float*)d_in[1];
  const float* b1  = (const float*)d_in[2];
  const float* W2  = (const float*)d_in[3];
  const float* b2  = (const float*)d_in[4];
  const float* Wd1 = (const float*)d_in[5];
  const float* bd1 = (const float*)d_in[6];
  const float* Wd2 = (const float*)d_in[7];
  const float* bd2 = (const float*)d_in[8];
  const float* cb  = (const float*)d_in[9];

  char* ws = (char*)d_ws;
  size_t off = 0;
  auto alloc = [&](size_t bytes) -> char* {
    char* p = ws + off;
    off += (bytes + 255) & ~(size_t)255;
    return p;
  };
  // transposed bf16 weights [N][K]
  __bf16* wW1t  = (__bf16*)alloc((size_t)EDIM * HDIM * 2);
  __bf16* wW2t  = (__bf16*)alloc((size_t)HDIM * HDIM * 2);
  __bf16* wWd1t = (__bf16*)alloc((size_t)HDIM * HDIM * 2);
  __bf16* wWd2t = (__bf16*)alloc((size_t)HDIM * EDIM * 2);
  __bf16* wCb   = (__bf16*)alloc((size_t)QLVL * KCODES * HDIM * 2);
  float*  cnorm      = (float*)alloc((size_t)QLVL * KCODES * 4);
  float*  reconPart  = (float*)alloc((size_t)1536 * 4);
  float*  commitPart = (float*)alloc((size_t)(BDIM / 128) * 4);
  __bf16* h1 = (__bf16*)alloc((size_t)BDIM * HDIM * 2);
  float*  zb = (float*) alloc((size_t)BDIM * HDIM * 4);
  __bf16* qz = (__bf16*)alloc((size_t)BDIM * HDIM * 2);
  __bf16* h2 = (__bf16*)alloc((size_t)BDIM * HDIM * 2);

  float* out     = (float*)d_out;
  float* idx_out = out + (size_t)BDIM * EDIM;
  float* tail    = idx_out + (size_t)BDIM * QLVL;

  // weight transpose+convert (tiny, L2-resident thereafter)
  cvtT_bf16_kernel<<<dim3(EDIM / 32, HDIM / 8), 256, 0, stream>>>(W1, wW1t, EDIM, HDIM);
  cvtT_bf16_kernel<<<dim3(HDIM / 32, HDIM / 8), 256, 0, stream>>>(W2, wW2t, HDIM, HDIM);
  cvtT_bf16_kernel<<<dim3(HDIM / 32, HDIM / 8), 256, 0, stream>>>(Wd1, wWd1t, HDIM, HDIM);
  cvtT_bf16_kernel<<<dim3(HDIM / 32, EDIM / 8), 256, 0, stream>>>(Wd2, wWd2t, HDIM, EDIM);
  cvt_bf16_kernel<<<(QLVL * KCODES * HDIM + 255) / 256, 256, 0, stream>>>(
      cb, wCb, QLVL * KCODES * HDIM);
  cnorm_kernel<<<QLVL * KCODES, 32, 0, stream>>>(cb, cnorm);

  dim3 blk(256);
  auto gsm = [](int kd) { return (size_t)(128 * (kd + 8) + 2 * 128 * AST) * 2 + 64; };
  // encoder
  gemm_kernel<float, 0, EDIM><<<dim3(HDIM / 128, BDIM / 128), blk, gsm(EDIM), stream>>>(
      x, wW1t, b1, h1, nullptr, nullptr, BDIM, HDIM);
  gemm_kernel<__bf16, 1, HDIM><<<dim3(HDIM / 128, BDIM / 128), blk, gsm(HDIM), stream>>>(
      h1, wW2t, b2, zb, nullptr, nullptr, BDIM, HDIM);
  // residual VQ
  size_t smem = ((size_t)128 * RSTRIDE + 8 * 272 + 128) * 4 + 128 * 4;
  rvq_kernel<<<BDIM / 128, blk, smem, stream>>>(zb, wCb, cb, cnorm, qz, idx_out,
                                                commitPart);
  // decoder
  gemm_kernel<__bf16, 0, HDIM><<<dim3(HDIM / 128, BDIM / 128), blk, gsm(HDIM), stream>>>(
      qz, wWd1t, bd1, h2, nullptr, nullptr, BDIM, HDIM);
  gemm_kernel<__bf16, 2, HDIM><<<dim3(EDIM / 128, BDIM / 128), blk, gsm(HDIM), stream>>>(
      h2, wWd2t, bd2, out, x, reconPart, BDIM, EDIM);
  // deterministic loss reduction
  finalize_kernel<<<1, 1, 0, stream>>>(reconPart, 3 * (BDIM / 128), commitPart,
                                       BDIM / 128, tail);
}
